// SelectiveStateSpace_82102594831008
// MI455X (gfx1250) — matrix operand order repaired and checked by reading, not compiled, not run on hardware
//
#include <hip/hip_runtime.h>
#include <hip/hip_bf16.h>
#include <math.h>

typedef __attribute__((ext_vector_type(16))) __bf16 v16bf;
typedef __attribute__((ext_vector_type(8)))  __bf16 v8bf;
typedef __attribute__((ext_vector_type(2)))  __bf16 v2bf;
typedef __attribute__((ext_vector_type(8)))  float  v8f;
typedef __attribute__((ext_vector_type(4)))  unsigned int uint4v;
typedef __attribute__((ext_vector_type(8)))  int    int8v;
typedef __attribute__((ext_vector_type(4)))  int    int4v;

#define B_   4
#define L_   4096
#define D_   1024
#define N_   64
#define K_   4
#define M_   (B_*L_)      // 16384 rows
#define P_   (2*N_)       // 128 proj cols
#define LC_  64           // scan chunk length
#define NC_  (L_/LC_)     // 64 chunks per sequence
#define ASTR 40           // LDS A-tile row stride in bf16 (80B, 16B-aligned)

static __device__ __forceinline__ float softplus_f(float z) {
  return (z > 20.f) ? z : log1pf(expf(z));
}

// ---------------------------------------------------------------------------
// Kernel 0: one-time weight prep: fp32 -> (hi,lo) bf16 split, transposed to
// [col][k] so WMMA B-fragments are single contiguous 32B vector loads.
// ---------------------------------------------------------------------------
__global__ __launch_bounds__(256)
void k_wprep(const float* __restrict__ Wp, const float* __restrict__ Ws,
             const float* __restrict__ Wo,
             __bf16* __restrict__ WpTh, __bf16* __restrict__ WpTl,
             __bf16* __restrict__ WsTh, __bf16* __restrict__ WsTl,
             __bf16* __restrict__ WoTh, __bf16* __restrict__ WoTl)
{
  const int i = blockIdx.x * 256 + threadIdx.x;
  if (i < D_*P_) {                       // Wp [k][col] -> [col][k]
    const int k = i >> 7, c = i & 127;
    const float w = Wp[i];
    const __bf16 h = (__bf16)w;
    WpTh[(size_t)c*D_ + k] = h;
    WpTl[(size_t)c*D_ + k] = (__bf16)(w - (float)h);
  } else if (i < D_*P_ + D_*N_) {        // Ws [k][col] -> [col][k]
    const int j = i - D_*P_;
    const int k = j >> 6, c = j & 63;
    const float w = Ws[j];
    const __bf16 h = (__bf16)w;
    WsTh[(size_t)c*D_ + k] = h;
    WsTl[(size_t)c*D_ + k] = (__bf16)(w - (float)h);
  } else {                               // Wo [n][d] -> [d][n]
    const int j = i - D_*P_ - D_*N_;
    const int n = j >> 10, d = j & 1023;
    const float w = Wo[j];
    const __bf16 h = (__bf16)w;
    WoTh[(size_t)d*N_ + n] = h;
    WoTl[(size_t)d*N_ + n] = (__bf16)(w - (float)h);
  }
}

// ---------------------------------------------------------------------------
// Kernel 1: fused causal depthwise conv (K=4) + proj GEMM (softplus/tanh
// epilogue) + u GEMM.  x tile arrives via the Tensor Data Mover; A tiles are
// built cooperatively in LDS as bf16 hi/lo; bf16x3-split WMMA accumulation.
// Block = 256 thr = 8 waves; wave: 16 rows x 192 cols; block: 128 rows.
// ---------------------------------------------------------------------------
__global__ __launch_bounds__(256)
void k_conv_proj(const float* __restrict__ x,  const float* __restrict__ cw,
                 const float* __restrict__ cb,
                 const __bf16* __restrict__ WpTh, const __bf16* __restrict__ WpTl,
                 const __bf16* __restrict__ WsTh, const __bf16* __restrict__ WsTl,
                 const float* __restrict__ bp, const float* __restrict__ bs,
                 float* __restrict__ Bt, float* __restrict__ Ct,
                 float* __restrict__ U)
{
  __shared__ float  xs[131][33];                 // x rows l0-3..l0+127 (fp32), TDM-padded stride 33
  __shared__ __bf16 ach_s[128][ASTR], acl_s[128][ASTR];   // conv  hi/lo A tile
  __shared__ __bf16 axh_s[128][ASTR], axl_s[128][ASTR];   // raw-x hi/lo A tile
  __shared__ float  wk[K_][32];
  __shared__ float  cbs[32];

  const int tid  = threadIdx.x;
  const int lane = tid & 31, wid = tid >> 5;
  const int mrow = lane & 15, half = lane >> 4;
  const int R0   = blockIdx.x * 128;             // one batch per block (128 | 4096)
  const int b    = R0 / L_;
  const int l0   = R0 % L_;

  v8f acc[12];
#pragma unroll
  for (int j = 0; j < 12; ++j) acc[j] = (v8f){0.f,0.f,0.f,0.f,0.f,0.f,0.f,0.f};

  for (int kt = 0; kt < D_; kt += 32) {
    __syncthreads();   // prior iteration done with xs / A tiles

    // ---- Tensor DMA: x[lstart .. lstart+rows) x 32ch  ->  xs (padded) ----
    if (wid == 0) {
      const int lstart = (l0 == 0) ? 0 : (l0 - 3);
      const int rows   = (l0 == 0) ? 128 : 131;
      const int r0     = (l0 == 0) ? 3 : 0;
      const unsigned long long ga =
          (unsigned long long)(uintptr_t)(x + ((size_t)(b*L_ + lstart))*D_ + kt);
      const unsigned int ldsa = (unsigned int)(uintptr_t)&xs[r0][0];
      const unsigned int td1  = 1u << 20;        // generous dim1 bound (no OOB in-tile)
      uint4v g0;
      g0.x = 1u;                                  // count=1 (user descriptor)
      g0.y = ldsa;                                // D#.lds_addr
      g0.z = (unsigned int)ga;                    // D#.global_addr[31:0]
      g0.w = (unsigned int)((ga >> 32) & 0x01FFFFFFu) | (2u << 30);  // addr[56:32] | type=2
      int8v g1;
      g1[0] = (int)((2u << 16) | (1u << 20) | (4u << 22));  // data_size=4B, pad_en, interval=32dw, amount=1dw
      g1[1] = (int)((D_ & 0xFFFFu) << 16);                  // tensor_dim0 lo16 (=1024)
      g1[2] = (int)((D_ >> 16) | ((td1 & 0xFFFFu) << 16));  // dim0 hi / dim1 lo
      g1[3] = (int)((td1 >> 16) | (32u << 16));             // dim1 hi / tile_dim0=32
      g1[4] = (int)(unsigned)rows;                          // tile_dim1=rows, tile_dim2=0
      g1[5] = (int)(unsigned)D_;                            // tensor_dim0_stride lo (elements)
      g1[6] = 0; g1[7] = 0;
      const int4v gz4 = (int4v){0,0,0,0};
      const int8v gz8 = (int8v){0,0,0,0,0,0,0,0};
      __builtin_amdgcn_tensor_load_to_lds(g0, g1, gz4, gz4, gz8, 0);
      __builtin_amdgcn_s_wait_tensorcnt(0);
    }
    if (l0 == 0 && tid < 3*32) xs[tid >> 5][tid & 31] = 0.f;  // causal left-pad
    if (tid < K_*32) wk[tid >> 5][tid & 31] = cw[(size_t)(tid >> 5)*D_ + kt + (tid & 31)];
    if (tid < 32)    cbs[tid] = cb[kt + tid];
    __syncthreads();

    // ---- cooperative conv + bf16 hi/lo split into LDS A tiles ----
    for (int i = tid; i < 128*16; i += 256) {
      const int r  = i >> 4;
      const int kp = (i & 15) * 2;
      float cv0 = cbs[kp], cv1 = cbs[kp+1];
      const float xv0 = xs[r+3][kp], xv1 = xs[r+3][kp+1];
#pragma unroll
      for (int j = 0; j < K_; ++j) {
        cv0 += wk[j][kp]   * xs[r+j][kp];
        cv1 += wk[j][kp+1] * xs[r+j][kp+1];
      }
      const __bf16 c0 = (__bf16)cv0, c1 = (__bf16)cv1;
      const __bf16 x0 = (__bf16)xv0, x1 = (__bf16)xv1;
      *(v2bf*)&ach_s[r][kp] = (v2bf){c0, c1};
      *(v2bf*)&acl_s[r][kp] = (v2bf){(__bf16)(cv0-(float)c0), (__bf16)(cv1-(float)c1)};
      *(v2bf*)&axh_s[r][kp] = (v2bf){x0, x1};
      *(v2bf*)&axl_s[r][kp] = (v2bf){(__bf16)(xv0-(float)x0), (__bf16)(xv1-(float)x1)};
    }
    __syncthreads();

    // ---- A fragments: two aligned 16B LDS vector loads each ----
    const int rr = wid*16 + mrow;
    v8bf c0 = *(const v8bf*)&ach_s[rr][16*half];
    v8bf c1 = *(const v8bf*)&ach_s[rr][16*half + 8];
    v8bf l0v = *(const v8bf*)&acl_s[rr][16*half];
    v8bf l1v = *(const v8bf*)&acl_s[rr][16*half + 8];
    v8bf x0v = *(const v8bf*)&axh_s[rr][16*half];
    v8bf x1v = *(const v8bf*)&axh_s[rr][16*half + 8];
    v8bf y0v = *(const v8bf*)&axl_s[rr][16*half];
    v8bf y1v = *(const v8bf*)&axl_s[rr][16*half + 8];
    v16bf ach, acl, axh, axl;
#pragma unroll
    for (int i = 0; i < 8; ++i) {
      ach[i] = c0[i];  ach[8+i] = c1[i];
      acl[i] = l0v[i]; acl[8+i] = l1v[i];
      axh[i] = x0v[i]; axh[8+i] = x1v[i];
      axl[i] = y0v[i]; axl[8+i] = y1v[i];
    }

    // ---- B fragments: single 32B global loads from transposed bf16 weights ----
#pragma unroll
    for (int j = 0; j < 8; ++j) {                 // proj cols 0..127
      const int col = j*16 + mrow;
      const v16bf bh = *(const v16bf*)&WpTh[(size_t)col*D_ + kt + 16*half];
      const v16bf bl = *(const v16bf*)&WpTl[(size_t)col*D_ + kt + 16*half];
      acc[j] = __builtin_amdgcn_wmma_f32_16x16x32_bf16(false, ach, false, bh, (short)0, acc[j], false, false);
      acc[j] = __builtin_amdgcn_wmma_f32_16x16x32_bf16(false, ach, false, bl, (short)0, acc[j], false, false);
      acc[j] = __builtin_amdgcn_wmma_f32_16x16x32_bf16(false, acl, false, bh, (short)0, acc[j], false, false);
    }
#pragma unroll
    for (int j = 0; j < 4; ++j) {                 // u cols 0..63
      const int col = j*16 + mrow;
      const v16bf bh = *(const v16bf*)&WsTh[(size_t)col*D_ + kt + 16*half];
      const v16bf bl = *(const v16bf*)&WsTl[(size_t)col*D_ + kt + 16*half];
      acc[8+j] = __builtin_amdgcn_wmma_f32_16x16x32_bf16(false, axh, false, bh, (short)0, acc[8+j], false, false);
      acc[8+j] = __builtin_amdgcn_wmma_f32_16x16x32_bf16(false, axh, false, bl, (short)0, acc[8+j], false, false);
      acc[8+j] = __builtin_amdgcn_wmma_f32_16x16x32_bf16(false, axl, false, bh, (short)0, acc[8+j], false, false);
    }
  }

  // ---- epilogue: bias + activations, write Bt/Ct/U [M, N] fp32 ----
#pragma unroll
  for (int j = 0; j < 8; ++j) {
    const int col = j*16 + (lane & 15);
    const float bias = bp[col];
#pragma unroll
    for (int v = 0; v < 8; ++v) {
      const int r = R0 + wid*16 + v + 8*half;
      const float z = acc[j][v] + bias;
      if (col < N_) Bt[(size_t)r*N_ + col]        = softplus_f(z);
      else          Ct[(size_t)r*N_ + (col - N_)] = tanhf(z);
    }
  }
#pragma unroll
  for (int j = 0; j < 4; ++j) {
    const int col = j*16 + (lane & 15);
    const float bias = bs[col];
#pragma unroll
    for (int v = 0; v < 8; ++v) {
      const int r = R0 + wid*16 + v + 8*half;
      U[(size_t)r*N_ + col] = acc[8+j][v] + bias;
    }
  }
}

// ---------------------------------------------------------------------------
// Kernel 2: chunk-local scan (constant-coefficient linear recurrence)
// ---------------------------------------------------------------------------
__global__ __launch_bounds__(64)
void k_scan_local(const float* __restrict__ U, const float* __restrict__ Bt,
                  const float* __restrict__ Ct,
                  const float* __restrict__ A_log, const float* __restrict__ dt_log,
                  float* __restrict__ Yloc, float* __restrict__ Carry)
{
  const int n  = threadIdx.x;
  const int ch = blockIdx.x;
  const int b  = blockIdx.y;
  const float dtv   = softplus_f(dt_log[n]);
  const float decay = 1.f - dtv * softplus_f(A_log[n]);
  const int t0 = ch * LC_;
  float s = 0.f;
  for (int i = 0; i < LC_; ++i) {
    const size_t idx = ((size_t)(b*L_ + t0 + i))*N_ + n;
    s = decay * s + dtv * Bt[idx] * U[idx];
    Yloc[idx] = Ct[idx] * s;
  }
  Carry[((size_t)b*NC_ + ch)*N_ + n] = s;
}

// ---------------------------------------------------------------------------
// Kernel 3: serial scan of chunk carries (tiny)
// ---------------------------------------------------------------------------
__global__ __launch_bounds__(256)
void k_scan_carry(const float* __restrict__ Carry,
                  const float* __restrict__ A_log, const float* __restrict__ dt_log,
                  float* __restrict__ Sprev)
{
  const int t = threadIdx.x;
  const int n = t & (N_-1), b = t >> 6;
  const float dtv   = softplus_f(dt_log[n]);
  const float decay = 1.f - dtv * softplus_f(A_log[n]);
  const float dLc   = powf(decay, (float)LC_);
  float s = 0.f;
  for (int ch = 0; ch < NC_; ++ch) {
    const size_t idx = ((size_t)b*NC_ + ch)*N_ + n;
    Sprev[idx] = s;
    s = dLc * s + Carry[idx];
  }
}

// ---------------------------------------------------------------------------
// Kernel 4: fixup y = yloc + C_t*decay^(i+1)*Sprev, emitted as bf16 hi/lo
// ---------------------------------------------------------------------------
__global__ __launch_bounds__(256)
void k_fixup(const float* __restrict__ Yloc, const float* __restrict__ Ct,
             const float* __restrict__ Sprev,
             const float* __restrict__ A_log, const float* __restrict__ dt_log,
             __bf16* __restrict__ Yh, __bf16* __restrict__ Yl)
{
  const size_t i = (size_t)blockIdx.x * 256 + threadIdx.x;  // over M_*N_ = 2^20
  const int n = (int)(i & (N_-1));
  const int t = (int)((i >> 6) & (L_-1));
  const int b = (int)(i >> 18);
  const float dtv   = softplus_f(dt_log[n]);
  const float decay = 1.f - dtv * softplus_f(A_log[n]);
  const int ch = t / LC_, o = t % LC_;
  const float p = powf(decay, (float)(o + 1));
  const float yv = Yloc[i] + Ct[i] * p * Sprev[((size_t)b*NC_ + ch)*N_ + n];
  const __bf16 h = (__bf16)yv;
  Yh[i] = h;
  Yl[i] = (__bf16)(yv - (float)h);
}

// ---------------------------------------------------------------------------
// Kernel 5: out = Y @ Wo + bo.  A from bf16 Yh/Yl via LDS; B from transposed
// bf16 WoT (single 32B global loads).  Block: 128 rows x 128 cols, 8 waves.
// ---------------------------------------------------------------------------
__global__ __launch_bounds__(256)
void k_out_gemm(const __bf16* __restrict__ Yh, const __bf16* __restrict__ Yl,
                const __bf16* __restrict__ WoTh, const __bf16* __restrict__ WoTl,
                const float* __restrict__ bo, float* __restrict__ Out)
{
  __shared__ __bf16 ysh[128][72];   // full-K (64) y hi tile, 144B stride (16B-aligned)
  __shared__ __bf16 ysl[128][72];

  const int tid  = threadIdx.x;
  const int lane = tid & 31, wid = tid >> 5;
  const int mrow = lane & 15, half = lane >> 4;
  const int R0 = blockIdx.x * 128;
  const int C0 = blockIdx.y * 128;

  for (int i = tid; i < 128*32; i += 256) {   // dword copies of 64 bf16/row
    const int r = i >> 5, dw = i & 31;
    ((unsigned int*)&ysh[r][0])[dw] = ((const unsigned int*)&Yh[(size_t)(R0 + r)*N_])[dw];
    ((unsigned int*)&ysl[r][0])[dw] = ((const unsigned int*)&Yl[(size_t)(R0 + r)*N_])[dw];
  }
  __syncthreads();

  v8f acc[8];
#pragma unroll
  for (int j = 0; j < 8; ++j) acc[j] = (v8f){0.f,0.f,0.f,0.f,0.f,0.f,0.f,0.f};

#pragma unroll
  for (int kt = 0; kt < N_; kt += 32) {
    const int rr = wid*16 + mrow;
    v8bf a0 = *(const v8bf*)&ysh[rr][kt + 16*half];
    v8bf a1 = *(const v8bf*)&ysh[rr][kt + 16*half + 8];
    v8bf b0 = *(const v8bf*)&ysl[rr][kt + 16*half];
    v8bf b1 = *(const v8bf*)&ysl[rr][kt + 16*half + 8];
    v16bf ah, al;
#pragma unroll
    for (int i = 0; i < 8; ++i) { ah[i]=a0[i]; ah[8+i]=a1[i]; al[i]=b0[i]; al[8+i]=b1[i]; }

#pragma unroll
    for (int j = 0; j < 8; ++j) {
      const int col = C0 + j*16 + mrow;
      const v16bf bh = *(const v16bf*)&WoTh[(size_t)col*N_ + kt + 16*half];
      const v16bf bl = *(const v16bf*)&WoTl[(size_t)col*N_ + kt + 16*half];
      acc[j] = __builtin_amdgcn_wmma_f32_16x16x32_bf16(false, ah, false, bh, (short)0, acc[j], false, false);
      acc[j] = __builtin_amdgcn_wmma_f32_16x16x32_bf16(false, ah, false, bl, (short)0, acc[j], false, false);
      acc[j] = __builtin_amdgcn_wmma_f32_16x16x32_bf16(false, al, false, bh, (short)0, acc[j], false, false);
    }
  }

#pragma unroll
  for (int j = 0; j < 8; ++j) {
    const int col = C0 + j*16 + (lane & 15);
    const float bias = bo[col];
#pragma unroll
    for (int v = 0; v < 8; ++v) {
      const int r = R0 + wid*16 + v + 8*half;
      Out[(size_t)r*D_ + col] = acc[j][v] + bias;
    }
  }
}

// ---------------------------------------------------------------------------
extern "C" void kernel_launch(void* const* d_in, const int* in_sizes, int n_in,
                              void* d_out, int out_size, void* d_ws, size_t ws_size,
                              hipStream_t stream) {
  const float* x      = (const float*)d_in[0];
  const float* cw     = (const float*)d_in[1];
  const float* cb     = (const float*)d_in[2];
  const float* Wp     = (const float*)d_in[3];
  const float* bp     = (const float*)d_in[4];
  const float* Ws     = (const float*)d_in[5];
  const float* bs     = (const float*)d_in[6];
  const float* A_log  = (const float*)d_in[7];
  const float* dt_log = (const float*)d_in[8];
  const float* Wo     = (const float*)d_in[9];
  const float* bo     = (const float*)d_in[10];
  float* out = (float*)d_out;

  // workspace layout
  char* w = (char*)d_ws;
  float*  Bt    = (float*) (w);                                    // 4 MB
  float*  Ct    = (float*) (w + ((size_t)4  << 20));               // 4 MB
  float*  U     = (float*) (w + ((size_t)8  << 20));               // 4 MB
  float*  Yloc  = (float*) (w + ((size_t)12 << 20));               // 4 MB
  __bf16* Yh    = (__bf16*)(w + ((size_t)16 << 20));               // 2 MB
  __bf16* Yl    = (__bf16*)(w + ((size_t)18 << 20));               // 2 MB
  float*  Carry = (float*) (w + ((size_t)20 << 20));               // 64 KB
  float*  Sprev = (float*) (w + ((size_t)20 << 20) + ((size_t)64  << 10));
  __bf16* WpTh  = (__bf16*)(w + ((size_t)21 << 20));               // 256 KB
  __bf16* WpTl  = (__bf16*)(w + ((size_t)21 << 20) + ((size_t)256 << 10));
  __bf16* WsTh  = (__bf16*)(w + ((size_t)21 << 20) + ((size_t)512 << 10));  // 128 KB
  __bf16* WsTl  = (__bf16*)(w + ((size_t)21 << 20) + ((size_t)640 << 10));
  __bf16* WoTh  = (__bf16*)(w + ((size_t)21 << 20) + ((size_t)768 << 10));  // 128 KB
  __bf16* WoTl  = (__bf16*)(w + ((size_t)21 << 20) + ((size_t)896 << 10));

  k_wprep<<<(D_*P_ + D_*N_ + N_*D_ + 255)/256, 256, 0, stream>>>(
      Wp, Ws, Wo, WpTh, WpTl, WsTh, WsTl, WoTh, WoTl);
  k_conv_proj<<<M_/128, 256, 0, stream>>>(
      x, cw, cb, WpTh, WpTl, WsTh, WsTl, bp, bs, Bt, Ct, U);
  k_scan_local<<<dim3(NC_, B_), 64, 0, stream>>>(U, Bt, Ct, A_log, dt_log, Yloc, Carry);
  k_scan_carry<<<1, 256, 0, stream>>>(Carry, A_log, dt_log, Sprev);
  k_fixup<<<(M_*N_)/256, 256, 0, stream>>>(Yloc, Ct, Sprev, A_log, dt_log, Yh, Yl);
  k_out_gemm<<<dim3(M_/128, D_/128), 256, 0, stream>>>(Yh, Yl, WoTh, WoTl, bo, out);
}
